// MaskMambaBlock_32126355374845
// MI455X (gfx1250) — compile-verified, matrix operand order checked by reading, not run on hardware
//
#include <hip/hip_runtime.h>
#include <hip/hip_bf16.h>
#include <stdint.h>

// ---------------------------------------------------------------------------
// MaskMambaBlock for MI455X (gfx1250, wave32, WMMA).
// B=4, T=2048, D=512, D_INNER=1024, D_STATE=16, D_CONV=4, DT_RANK=32
// ---------------------------------------------------------------------------

#define B_    4
#define T_    2048
#define D_    512
#define DI_   1024
#define DS_   16
#define DC_   4
#define DR_   32
#define M_    (B_ * T_)        // 8192 rows

typedef _Float16 v16h __attribute__((ext_vector_type(16)));
typedef _Float16 v8h  __attribute__((ext_vector_type(8)));
typedef float    v8f  __attribute__((ext_vector_type(8)));

// ---------------------------------------------------------------------------
// helpers
// ---------------------------------------------------------------------------
__device__ __forceinline__ float silu_f(float x) {
    return x * (1.0f / (1.0f + __expf(-x)));
}

// Load one 16x32 f16 WMMA operand fragment for this lane.
// p must already point at  base + (tile_row + (lane&15))*ld + kbase + gsel
// where gsel = (lane<16)?0:8.  Covers K offsets {0..7, 16..23} (lanes 0-15)
// or {8..15, 24..31} (lanes 16-31) per the CDNA5 16-bit A/B VGPR layout.
__device__ __forceinline__ v16h load_frag(const _Float16* __restrict__ p) {
    v8h lo = *reinterpret_cast<const v8h*>(p);
    v8h hi = *reinterpret_cast<const v8h*>(p + 16);
    v16h r;
#pragma unroll
    for (int i = 0; i < 8; ++i) { r[i] = lo[i]; r[i + 8] = hi[i]; }
    return r;
}

// ---------------------------------------------------------------------------
// elementwise f32 -> f16 cast (weight conversion)
// ---------------------------------------------------------------------------
__global__ void cast_f32_f16_kernel(const float* __restrict__ src,
                                    _Float16* __restrict__ dst, int n) {
    int i = blockIdx.x * blockDim.x + threadIdx.x;
    if (i < n) dst[i] = (_Float16)src[i];
}

// ---------------------------------------------------------------------------
// LayerNorm over D for each (b,t) row of x (layout B,D,T), emit f16 xn (M x D)
// ---------------------------------------------------------------------------
__global__ __launch_bounds__(256)
void ln_kernel(const float* __restrict__ x, const float* __restrict__ gamma,
               const float* __restrict__ beta, _Float16* __restrict__ xn) {
    const int row = blockIdx.x;            // 0..M-1
    const int b   = row >> 11;             // /T
    const int t   = row & (T_ - 1);
    const int tid = threadIdx.x;
    const float* xb = x + (size_t)b * D_ * T_;

    __shared__ float red[256];

    float v0 = xb[(size_t)tid * T_ + t];
    float v1 = xb[(size_t)(tid + 256) * T_ + t];

    red[tid] = v0 + v1;
    __syncthreads();
#pragma unroll
    for (int off = 128; off > 0; off >>= 1) {
        if (tid < off) red[tid] += red[tid + off];
        __syncthreads();
    }
    const float mu = red[0] * (1.0f / (float)D_);
    __syncthreads();

    const float d0 = v0 - mu, d1 = v1 - mu;
    red[tid] = d0 * d0 + d1 * d1;
    __syncthreads();
#pragma unroll
    for (int off = 128; off > 0; off >>= 1) {
        if (tid < off) red[tid] += red[tid + off];
        __syncthreads();
    }
    const float var = red[0] * (1.0f / (float)D_);
    const float rs  = rsqrtf(var + 1e-5f);

    xn[(size_t)row * D_ + tid]       = (_Float16)(d0 * rs * gamma[tid]       + beta[tid]);
    xn[(size_t)row * D_ + tid + 256] = (_Float16)(d1 * rs * gamma[tid + 256] + beta[tid + 256]);
}

// ---------------------------------------------------------------------------
// WMMA GEMM:  C[M,N] = A[M,K] * W[N,K]^T   (A,W f16 row-major)
// Block = 256 threads = 8 waves; block tile 128(M) x 64(N); wave tile 32x32.
// EPI 0: in_proj epilogue  (n<1024 -> u f32 ; n>=1024 -> z f16)
// EPI 1: out_proj epilogue (residual + scale*mask*y, transpose to (B,D,T))
// ---------------------------------------------------------------------------
template <int KDIM, int EPI>
__global__ __launch_bounds__(256)
void gemm_wmma_kernel(const _Float16* __restrict__ A,
                      const _Float16* __restrict__ W,
                      float* __restrict__ u_out,          // EPI 0
                      _Float16* __restrict__ z_out,       // EPI 0
                      const float* __restrict__ x_res,    // EPI 1
                      const unsigned char* __restrict__ mask, // EPI 1
                      const float* __restrict__ scale,    // EPI 1
                      float* __restrict__ final_out) {    // EPI 1
    const int lane  = threadIdx.x & 31;
    const int wave  = threadIdx.x >> 5;       // 0..7
    const int m0    = blockIdx.y * 128 + (wave >> 1) * 32;
    const int n0    = blockIdx.x * 64  + (wave & 1) * 32;
    const int r15   = lane & 15;
    const int gsel  = (lane < 16) ? 0 : 8;

    const _Float16* pa0 = A + (size_t)(m0 + r15) * KDIM + gsel;
    const _Float16* pa1 = pa0 + (size_t)16 * KDIM;
    const _Float16* pb0 = W + (size_t)(n0 + r15) * KDIM + gsel;
    const _Float16* pb1 = pb0 + (size_t)16 * KDIM;

    v8f acc[2][2];
#pragma unroll
    for (int i = 0; i < 2; ++i)
#pragma unroll
        for (int j = 0; j < 2; ++j) acc[i][j] = (v8f){};

    for (int k = 0; k < KDIM; k += 32) {
        // hint the next k-slab into cache (speculative, OOB-safe)
        __builtin_prefetch(pa0 + k + 128, 0, 1);
        __builtin_prefetch(pb0 + k + 128, 0, 1);

        v16h a0 = load_frag(pa0 + k);
        v16h a1 = load_frag(pa1 + k);
        v16h b0 = load_frag(pb0 + k);
        v16h b1 = load_frag(pb1 + k);

        acc[0][0] = __builtin_amdgcn_wmma_f32_16x16x32_f16(false, a0, false, b0,
                                                           (short)0, acc[0][0], false, false);
        acc[0][1] = __builtin_amdgcn_wmma_f32_16x16x32_f16(false, a0, false, b1,
                                                           (short)0, acc[0][1], false, false);
        acc[1][0] = __builtin_amdgcn_wmma_f32_16x16x32_f16(false, a1, false, b0,
                                                           (short)0, acc[1][0], false, false);
        acc[1][1] = __builtin_amdgcn_wmma_f32_16x16x32_f16(false, a1, false, b1,
                                                           (short)0, acc[1][1], false, false);
    }

    // epilogue: element r of v8f is (M = tile_m + r + gsel, N = tile_n + r15)
#pragma unroll
    for (int i = 0; i < 2; ++i) {
#pragma unroll
        for (int j = 0; j < 2; ++j) {
            const int ntile = n0 + 16 * j;
            const int nn    = ntile + r15;
#pragma unroll
            for (int r = 0; r < 8; ++r) {
                const int   mm = m0 + 16 * i + r + gsel;
                const float v  = acc[i][j][r];
                if (EPI == 0) {
                    if (ntile < DI_)
                        u_out[(size_t)mm * DI_ + nn] = v;
                    else
                        z_out[(size_t)mm * DI_ + (nn - DI_)] = (_Float16)v;
                } else {
                    const int b = mm >> 11;
                    const int t = mm & (T_ - 1);
                    const float mv = mask[(size_t)b * T_ + t] ? 1.0f : 0.0f;
                    const size_t idx = ((size_t)b * D_ + nn) * T_ + t;
                    final_out[idx] = x_res[idx] + scale[nn] * (v * mv);
                }
            }
        }
    }
}

// ---------------------------------------------------------------------------
// depthwise causal conv (kernel 4) + bias + SiLU.  u layout: (M x DI) row-major.
// ---------------------------------------------------------------------------
__global__ __launch_bounds__(256)
void conv_silu_kernel(const float* __restrict__ u,
                      const float* __restrict__ cw,
                      const float* __restrict__ cb,
                      float* __restrict__ uc) {
    const size_t i = (size_t)blockIdx.x * 256 + threadIdx.x;  // over M*DI
    const int d = (int)(i & (DI_ - 1));
    const int m = (int)(i >> 10);
    const int b = m >> 11;
    const int t = m & (T_ - 1);

    float acc = cb[d];
#pragma unroll
    for (int j = 0; j < DC_; ++j) {
        const int tt = t - (DC_ - 1) + j;
        if (tt >= 0)
            acc += cw[d * DC_ + j] * u[((size_t)(b * T_ + tt)) * DI_ + d];
    }
    uc[i] = silu_f(acc);
}

// ---------------------------------------------------------------------------
// x_proj: xdbl[m,e] = sum_k uc[m,k] * xw[e,k]  (e < 64, k < 1024)
// one block per row m; row staged in LDS.
// ---------------------------------------------------------------------------
__global__ __launch_bounds__(64)
void xproj_kernel(const float* __restrict__ uc,
                  const float* __restrict__ xw,
                  float* __restrict__ xdbl) {
    __shared__ float row[DI_];
    const int m = blockIdx.x;
    for (int i = threadIdx.x; i < DI_; i += 64)
        row[i] = uc[(size_t)m * DI_ + i];
    __syncthreads();

    const int e = threadIdx.x;     // 0..63
    const float* w = xw + (size_t)e * DI_;
    float acc = 0.0f;
#pragma unroll 8
    for (int k = 0; k < DI_; ++k) acc += row[k] * w[k];
    xdbl[(size_t)m * 64 + e] = acc;
}

// ---------------------------------------------------------------------------
// dt_proj + softplus:  delta[m,d] = softplus( sum_r xdbl[m,r]*dtw[d,r] + dtb[d] )
// ---------------------------------------------------------------------------
__global__ __launch_bounds__(256)
void delta_kernel(const float* __restrict__ xdbl,
                  const float* __restrict__ dtw,
                  const float* __restrict__ dtb,
                  float* __restrict__ delta) {
    const size_t i = (size_t)blockIdx.x * 256 + threadIdx.x;  // over M*DI
    const int d = (int)(i & (DI_ - 1));
    const int m = (int)(i >> 10);

    const float* dt = xdbl + (size_t)m * 64;   // first DR_ cols are dt
    const float* w  = dtw + (size_t)d * DR_;
    float acc = dtb[d];
#pragma unroll
    for (int r = 0; r < DR_; ++r) acc += dt[r] * w[r];
    delta[i] = (acc > 20.0f) ? acc : log1pf(__expf(acc));
}

// ---------------------------------------------------------------------------
// selective scan: sequential over T.  One thread per (b, d) channel, 16 states
// in registers; B/C chunks staged in LDS (shared across all d in the block).
// Fused epilogue: yg = (y + u*Dp) * silu(z), stored f16 for out_proj GEMM.
// ---------------------------------------------------------------------------
__global__ __launch_bounds__(256)
void scan_kernel(const float* __restrict__ xdbl,
                 const float* __restrict__ delta,
                 const float* __restrict__ uc,
                 const _Float16* __restrict__ zf,
                 const float* __restrict__ A_log,
                 const float* __restrict__ Dp,
                 _Float16* __restrict__ yg) {
    const int b = blockIdx.x >> 2;                       // 4 blocks per batch
    const int d = (blockIdx.x & 3) * 256 + threadIdx.x;  // 0..1023

    float Arow[DS_], h[DS_];
#pragma unroll
    for (int s = 0; s < DS_; ++s) {
        Arow[s] = -__expf(A_log[(size_t)d * DS_ + s]);
        h[s] = 0.0f;
    }
    const float dp = Dp[d];

    const int TC = 32;                      // timesteps per LDS chunk
    __shared__ float Bc[TC][DS_];
    __shared__ float Cc[TC][DS_];

    for (int t0 = 0; t0 < T_; t0 += TC) {
        __syncthreads();
        // cooperative load of B (cols 32..47) and C (cols 48..63) for TC steps
        for (int i = threadIdx.x; i < TC * 32; i += 256) {
            const int tt = i >> 5;
            const int c  = i & 31;
            const float v = xdbl[((size_t)(b * T_ + t0 + tt)) * 64 + 32 + c];
            if (c < DS_) Bc[tt][c] = v;
            else         Cc[tt][c - DS_] = v;
        }
        __syncthreads();

        for (int tl = 0; tl < TC; ++tl) {
            const size_t m  = (size_t)b * T_ + t0 + tl;
            const float dlt = delta[m * DI_ + d];
            const float ut  = uc[m * DI_ + d];
            float y = 0.0f;
#pragma unroll
            for (int s = 0; s < DS_; ++s) {
                const float dA = __expf(dlt * Arow[s]);
                h[s] = dA * h[s] + (dlt * Bc[tl][s]) * ut;
                y += h[s] * Cc[tl][s];
            }
            const float zv = (float)zf[m * DI_ + d];
            yg[m * DI_ + d] = (_Float16)((y + ut * dp) * silu_f(zv));
        }
    }
}

// ---------------------------------------------------------------------------
// launch
// ---------------------------------------------------------------------------
extern "C" void kernel_launch(void* const* d_in, const int* in_sizes, int n_in,
                              void* d_out, int out_size, void* d_ws, size_t ws_size,
                              hipStream_t stream) {
    (void)in_sizes; (void)n_in; (void)out_size; (void)ws_size;

    const float*         x       = (const float*)d_in[0];          // (B,D,T)
    const unsigned char* mask    = (const unsigned char*)d_in[1];  // (B,T) bool
    const float*         ln_g    = (const float*)d_in[2];
    const float*         ln_b    = (const float*)d_in[3];
    const float*         w_in    = (const float*)d_in[4];          // (2048,512)
    const float*         conv_w  = (const float*)d_in[5];          // (1024,4)
    const float*         conv_b  = (const float*)d_in[6];
    const float*         x_projw = (const float*)d_in[7];          // (64,1024)
    const float*         dt_w    = (const float*)d_in[8];          // (1024,32)
    const float*         dt_b    = (const float*)d_in[9];
    const float*         A_log   = (const float*)d_in[10];         // (1024,16)
    const float*         Dp      = (const float*)d_in[11];
    const float*         w_out   = (const float*)d_in[12];         // (512,1024)
    const float*         scale   = (const float*)d_in[13];         // (512)
    float*               out     = (float*)d_out;                  // (B,D,T)

    // -------- workspace carve-up --------
    char* ws = (char*)d_ws;
    _Float16* xn_h   = (_Float16*)ws; ws += (size_t)M_ * D_ * sizeof(_Float16);   //  8 MB
    _Float16* w_in_h = (_Float16*)ws; ws += (size_t)2 * DI_ * D_ * sizeof(_Float16); // 2 MB
    _Float16* w_out_h= (_Float16*)ws; ws += (size_t)D_ * DI_ * sizeof(_Float16);  //  1 MB
    float*    u_f    = (float*)ws;    ws += (size_t)M_ * DI_ * sizeof(float);     // 32 MB
    _Float16* z_h    = (_Float16*)ws; ws += (size_t)M_ * DI_ * sizeof(_Float16);  // 16 MB
    float*    uc_f   = (float*)ws;    ws += (size_t)M_ * DI_ * sizeof(float);     // 32 MB
    float*    xdbl_f = (float*)ws;    ws += (size_t)M_ * 64 * sizeof(float);      //  2 MB
    float*    delta_f= (float*)ws;    ws += (size_t)M_ * DI_ * sizeof(float);     // 32 MB
    _Float16* yg_h   = (_Float16*)ws; ws += (size_t)M_ * DI_ * sizeof(_Float16);  // 16 MB

    // -------- weight f16 conversion --------
    {
        int n1 = 2 * DI_ * D_;
        cast_f32_f16_kernel<<<(n1 + 255) / 256, 256, 0, stream>>>(w_in, w_in_h, n1);
        int n2 = D_ * DI_;
        cast_f32_f16_kernel<<<(n2 + 255) / 256, 256, 0, stream>>>(w_out, w_out_h, n2);
    }

    // -------- LayerNorm + cast to f16 --------
    ln_kernel<<<M_, 256, 0, stream>>>(x, ln_g, ln_b, xn_h);

    // -------- in_proj GEMM (M=8192, N=2048, K=512) with u/z split epilogue ----
    gemm_wmma_kernel<D_, 0><<<dim3((2 * DI_) / 64, M_ / 128), 256, 0, stream>>>(
        xn_h, w_in_h, u_f, z_h, nullptr, nullptr, nullptr, nullptr);

    // -------- depthwise conv4 + SiLU --------
    conv_silu_kernel<<<(M_ * DI_) / 256, 256, 0, stream>>>(u_f, conv_w, conv_b, uc_f);

    // -------- x_proj (M x 64, K=1024) --------
    xproj_kernel<<<M_, 64, 0, stream>>>(uc_f, x_projw, xdbl_f);

    // -------- dt_proj + softplus --------
    delta_kernel<<<(M_ * DI_) / 256, 256, 0, stream>>>(xdbl_f, dt_w, dt_b, delta_f);

    // -------- selective scan + gating, emit f16 --------
    scan_kernel<<<B_ * 4, 256, 0, stream>>>(xdbl_f, delta_f, uc_f, z_h, A_log, Dp, yg_h);

    // -------- out_proj GEMM (M=8192, N=512, K=1024) + residual/scale/mask ----
    gemm_wmma_kernel<DI_, 1><<<dim3(D_ / 64, M_ / 128), 256, 0, stream>>>(
        yg_h, w_out_h, nullptr, nullptr, x, mask, scale, out);
}
